// ParallelINN_43757126812048
// MI455X (gfx1250) — matrix operand order repaired and checked by reading, not compiled, not run on hardware
//
#include <hip/hip_runtime.h>
#include <hip/hip_bf16.h>
#include <cstdint>

// ---------------------------------------------------------------------------
// ParallelINN (Mamba + 4-neuron attention) for MI455X / gfx1250.
// GEMMs: v_wmma_f32_16x16x32_bf16 with async-LDS (ASYNCcnt) A-tile staging and
// Tensor Data Mover (TENSORcnt) B-tile staging.
// ---------------------------------------------------------------------------

typedef __bf16 bf16;
typedef bf16  v16bf __attribute__((ext_vector_type(16)));
typedef float v8f   __attribute__((ext_vector_type(8)));
typedef unsigned int u32x4 __attribute__((ext_vector_type(4)));
typedef int i32x4 __attribute__((ext_vector_type(4)));
typedef int i32x8 __attribute__((ext_vector_type(8)));

#define D_MODEL   512
#define N_NEURONS 4
#define N_HEADS   4
#define D_INNER   1024
#define D_STATE   16
#define DT_RANK   32
#define BATCH     2
#define SEQ       512
#define NTOK      (BATCH * N_NEURONS * SEQ)   // 4096 token rows in (b,n,l) order
#define VOCAB     32000
#define CEIL(a, b) (((a) + (b) - 1) / (b))

__device__ __forceinline__ bf16 f2bf(float f) {
  union { float f; uint32_t u; } v; v.f = f;
  uint32_t r = v.u + 0x7fffu + ((v.u >> 16) & 1u);   // round-to-nearest-even
  uint16_t h = (uint16_t)(r >> 16);
  return __builtin_bit_cast(bf16, h);
}
__device__ __forceinline__ float siluf(float x) { return x / (1.f + __expf(-x)); }

// Per-lane async global->LDS copy of 16 bytes (GLOBAL_LOAD_ASYNC_TO_LDS_B128,
// tracked with ASYNCcnt; LDS address = low 32 bits of generic address).
__device__ __forceinline__ void async_ld_b128(uint32_t lds_off, const void* gaddr) {
  asm volatile("global_load_async_to_lds_b128 %0, %1, off"
               :: "v"(lds_off), "v"(gaddr) : "memory");
}
__device__ __forceinline__ void wait_asynccnt0() {
  asm volatile("s_wait_asynccnt 0" ::: "memory");
}

// ---------------------------------------------------------------------------
// Generic WMMA GEMM:  C[M,N] (+)= A[M,K] * W[N,K]^T   (A,W bf16 row-major, C f32)
// Block = 128 threads (4 waves). Block tile 64x64, K-step 32.
// ---------------------------------------------------------------------------
__global__ void __launch_bounds__(128) gemm_bf16_wmma(
    const bf16* __restrict__ A, const bf16* __restrict__ W,
    float* __restrict__ C, int M, int N, int K, int accum)
{
  __shared__ __align__(16) bf16 lA[64][32];
  __shared__ __align__(16) bf16 lB[64][32];

  const int m0 = blockIdx.y * 64;
  const int n0 = blockIdx.x * 64;
  const int wv = threadIdx.x >> 5;
  const int ln = threadIdx.x & 31;
  const int lnlo = ln & 15;
  const int lnhi = ln >> 4;

  const uint32_t ldsA = (uint32_t)(uintptr_t)&lA[0][0];
  const uint32_t ldsB = (uint32_t)(uintptr_t)&lB[0][0];

  v8f acc[4];
#pragma unroll
  for (int t = 0; t < 4; ++t)
#pragma unroll
    for (int j = 0; j < 8; ++j) acc[t][j] = 0.f;

  for (int k0 = 0; k0 < K; k0 += 32) {
    // Prefetch next K-tile of A into cache (global_prefetch_b8).
    if (k0 + 32 < K) {
      int pr = threadIdx.x >> 1;
      int pq = (threadIdx.x & 1) << 4;
      if (m0 + pr < M) __builtin_prefetch(A + (size_t)(m0 + pr) * K + k0 + 32 + pq, 0, 1);
    }

    // ---- A tile: 64x32 bf16 via per-lane async global->LDS b128 copies ----
#pragma unroll
    for (int i = 0; i < 2; ++i) {
      int c   = threadIdx.x + i * 128;   // 0..255 chunks of 16B
      int row = c >> 2;
      int kq  = c & 3;
      if (m0 + row < M)
        async_ld_b128(ldsA + (uint32_t)(row * 64 + kq * 16),
                      A + (size_t)(m0 + row) * K + k0 + kq * 8);
    }

    // ---- B tile: 32(K) x 64(N-rows) bf16 via Tensor Data Mover ----
#if __has_builtin(__builtin_amdgcn_tensor_load_to_lds)
    if (wv == 0) {
      uint64_t ga = (uint64_t)(uintptr_t)(W + (size_t)n0 * K + k0);
      uint32_t rows_left = (uint32_t)(N - n0);
      u32x4 g0 = {0u, 0u, 0u, 0u};
      g0[0] = 1u;                                  // count=1, user descriptor
      g0[1] = ldsB;                                // lds_addr
      g0[2] = (uint32_t)ga;                        // global_addr[31:0]
      g0[3] = (uint32_t)(ga >> 32) | (2u << 30);   // global_addr[56:32] | type=2
      i32x8 g1 = {0, 0, 0, 0, 0, 0, 0, 0};
      g1[0] = 0x00010000;                          // data_size=1 (2 bytes), no multicast
      g1[1] = (int)(((uint32_t)K & 0xffffu) << 16);            // tensor_dim0 lo16
      g1[2] = (int)(((uint32_t)K >> 16) | ((rows_left & 0xffffu) << 16)); // dim0 hi | dim1 lo
      g1[3] = (int)((rows_left >> 16) | (32u << 16));          // dim1 hi | tile_dim0=32
      g1[4] = 64;                                  // tile_dim1=64, tile_dim2=0
      g1[5] = (int)K;                              // tensor_dim0_stride lo32
      i32x4 g2 = {0, 0, 0, 0};
      i32x4 g3 = {0, 0, 0, 0};
#if defined(__clang_major__) && (__clang_major__ >= 23)
      i32x8 g4 = {0, 0, 0, 0, 0, 0, 0, 0};
      __builtin_amdgcn_tensor_load_to_lds(g0, g1, g2, g3, g4, 0);
#else
      __builtin_amdgcn_tensor_load_to_lds(g0, g1, g2, g3, 0);
#endif
      __builtin_amdgcn_s_wait_tensorcnt(0);
    }
#else
    // Fallback: per-lane async copies for B as well.
#pragma unroll
    for (int i = 0; i < 2; ++i) {
      int c   = threadIdx.x + i * 128;
      int row = c >> 2;
      int kq  = c & 3;
      if (n0 + row < N)
        async_ld_b128(ldsB + (uint32_t)(row * 64 + kq * 16),
                      W + (size_t)(n0 + row) * K + k0 + kq * 8);
    }
#endif
    wait_asynccnt0();
    __syncthreads();

    // A fragment (16x32 bf16, ISA 7.12.2 layout):
    //   lanes 0-15: K 0-7 / 16-23; lanes 16-31: K 8-15 / 24-31  (two b128 LDS loads)
    v16bf af;
    const int mrow = (wv << 4) + lnlo;
#pragma unroll
    for (int h = 0; h < 16; ++h) {
      int kk = ((h & 8) << 1) | (lnhi << 3) | (h & 7);
      af[h] = lA[mrow][kk];
    }
    // B fragments (32x16): lane holds col N=ln&15; lanes 0-15 K 0-7 / 16-23, lanes 16-31 K 8-15 / 24-31.
#pragma unroll
    for (int nt = 0; nt < 4; ++nt) {
      v16bf bfr;
      const int col = (nt << 4) + lnlo;
      const int kb  = lnhi << 4;
#pragma unroll
      for (int h = 0; h < 16; ++h) bfr[h] = lB[col][((h & 8) << 1) | (lnhi << 3) | (h & 7)];
      acc[nt] = __builtin_amdgcn_wmma_f32_16x16x32_bf16(
          false, af, false, bfr, (short)0, acc[nt], false, false);
    }
    __syncthreads();
  }

  // C layout: VGPR j -> M = j + 8*lnhi, N = lnlo.
  const int rofs = lnhi << 3;
#pragma unroll
  for (int nt = 0; nt < 4; ++nt) {
#pragma unroll
    for (int j = 0; j < 8; ++j) {
      int r  = m0 + (wv << 4) + rofs + j;
      int cc = n0 + (nt << 4) + lnlo;
      if (r < M && cc < N) {
        size_t idx = (size_t)r * N + cc;
        C[idx] = accum ? (C[idx] + acc[nt][j]) : acc[nt][j];
      }
    }
  }
}

// ---------------------------------------------------------------------------
// Elementwise / small kernels
// ---------------------------------------------------------------------------
__global__ void f2bf_kernel(const float* __restrict__ src, bf16* __restrict__ dst, int n) {
  int i = blockIdx.x * blockDim.x + threadIdx.x;
  if (i < n) dst[i] = f2bf(src[i]);
}

// x[b,n,l,d] = emb[ids[b,l], d]  (broadcast over n)
__global__ void embed_kernel(const int* __restrict__ ids, const float* __restrict__ emb,
                             float* __restrict__ x) {
  size_t i = (size_t)blockIdx.x * blockDim.x + threadIdx.x;
  if (i >= (size_t)NTOK * D_MODEL) return;
  int d = (int)(i & 511);
  size_t t = i >> 9;
  int l = (int)(t & 511); t >>= 9;
  t >>= 2;                     // skip n
  int b = (int)t;
  x[i] = emb[(size_t)ids[b * SEQ + l] * D_MODEL + d];
}

// Depthwise causal conv (k=4) + bias + SiLU over the x_in half of xr.
__global__ void __launch_bounds__(256) conv_silu_kernel(
    const float* __restrict__ xr, const float* __restrict__ conv_w,
    const float* __restrict__ conv_b, float* __restrict__ xconv,
    bf16* __restrict__ xconv_b)
{
  int c  = blockIdx.x * blockDim.x + threadIdx.x;   // 0..1023
  int bn = blockIdx.y;                               // 0..7
  int ch = (bn & 3) * D_INNER + c;                   // channel in [0,4096)
  float w0 = conv_w[ch * 4 + 0], w1 = conv_w[ch * 4 + 1];
  float w2 = conv_w[ch * 4 + 2], w3 = conv_w[ch * 4 + 3];
  float bias = conv_b[ch];
  const float* base = xr + (size_t)bn * SEQ * (2 * D_INNER) + c;
  float* ob  = xconv   + (size_t)bn * SEQ * D_INNER + c;
  bf16*  obb = xconv_b + (size_t)bn * SEQ * D_INNER + c;
  float xm3 = 0.f, xm2 = 0.f, xm1 = 0.f;
  for (int l = 0; l < SEQ; ++l) {
    float x0 = base[(size_t)l * (2 * D_INNER)];
    float a  = w0 * xm3 + w1 * xm2 + w2 * xm1 + w3 * x0 + bias;
    float s  = siluf(a);
    ob[(size_t)l * D_INNER]  = s;
    obb[(size_t)l * D_INNER] = f2bf(s);
    xm3 = xm2; xm2 = xm1; xm1 = x0;
  }
}

// Extract dt columns (first 32 of dtBC's 64) to bf16 for dt_proj GEMM.
__global__ void dtcols_kernel(const float* __restrict__ dtbc, bf16* __restrict__ out, int n) {
  int i = blockIdx.x * blockDim.x + threadIdx.x;
  if (i >= n) return;
  int r = i >> 5, j = i & 31;
  out[i] = f2bf(dtbc[(size_t)r * 64 + j]);
}

// dt = softplus(dtproj + bias)
__global__ void softplus_kernel(float* __restrict__ dt, const float* __restrict__ bias, int n) {
  int i = blockIdx.x * blockDim.x + threadIdx.x;
  if (i >= n) return;
  float v = dt[i] + bias[i & (D_INNER - 1)];
  dt[i] = (v > 20.f) ? v : log1pf(__expf(v));
}

__global__ void negexp_kernel(const float* __restrict__ a_log, float* __restrict__ aneg, int n) {
  int i = blockIdx.x * blockDim.x + threadIdx.x;
  if (i < n) aneg[i] = -__expf(a_log[i]);
}

// Selective scan: one thread per (bn, channel), 16-state recurrence in registers.
__global__ void __launch_bounds__(256) ssm_scan_kernel(
    const float* __restrict__ dt, const float* __restrict__ xconv,
    const float* __restrict__ dtbc, const float* __restrict__ aneg,
    const float* __restrict__ dskip, float* __restrict__ y)
{
  int c  = blockIdx.x * blockDim.x + threadIdx.x;   // 0..1023
  int bn = blockIdx.y;                               // 0..7
  float h[D_STATE], Av[D_STATE];
#pragma unroll
  for (int s = 0; s < D_STATE; ++s) { h[s] = 0.f; Av[s] = aneg[c * D_STATE + s]; }
  float dsk = dskip[c];
  const float* dtp = dt    + (size_t)bn * SEQ * D_INNER + c;
  const float* xp  = xconv + (size_t)bn * SEQ * D_INNER + c;
  const float* bc  = dtbc  + (size_t)bn * SEQ * 64;
  float* yp        = y     + (size_t)bn * SEQ * D_INNER + c;
  for (int l = 0; l < SEQ; ++l) {
    float dtv = dtp[(size_t)l * D_INNER];
    float xv  = xp[(size_t)l * D_INNER];
    float dbx = dtv * xv;
    const float* bcl = bc + (size_t)l * 64;
    float acc = 0.f;
#pragma unroll
    for (int s = 0; s < D_STATE; ++s) {
      h[s] = __expf(dtv * Av[s]) * h[s] + dbx * bcl[32 + s];
      acc += h[s] * bcl[48 + s];
    }
    yp[(size_t)l * D_INNER] = acc + xv * dsk;
  }
}

// yb = bf16( y * silu(res) ), res = second half of xr rows.
__global__ void gate_kernel(const float* __restrict__ y, const float* __restrict__ xr,
                            bf16* __restrict__ yb, int n) {
  int i = blockIdx.x * blockDim.x + threadIdx.x;
  if (i >= n) return;
  int r = i >> 10, c = i & (D_INNER - 1);
  float res = xr[(size_t)r * (2 * D_INNER) + D_INNER + c];
  yb[i] = f2bf(y[i] * siluf(res));
}

// Tiny attention across the 4 neurons for each (b, l, head, query-neuron).
__global__ void __launch_bounds__(256) attn_kernel(
    const float* __restrict__ qkv, const float* __restrict__ b_in,
    float* __restrict__ ao)
{
  int tid = blockIdx.x * blockDim.x + threadIdx.x;
  if (tid >= BATCH * SEQ * N_HEADS * N_NEURONS) return;
  int nq  = tid & 3;
  int h   = (tid >> 2) & 3;
  int tok = tid >> 4;                  // b*512 + l
  int b = tok >> 9, l = tok & 511;
  size_t rows[4];
#pragma unroll
  for (int n = 0; n < 4; ++n) rows[n] = (size_t)(b * 4 + n) * SEQ + l;
  const int hb = h * 128;
  float sc[4];
  for (int nk = 0; nk < 4; ++nk) {
    const float* qp = qkv + rows[nq] * 1536 + hb;
    const float* kp = qkv + rows[nk] * 1536 + 512 + hb;
    float s = 0.f;
    for (int d = 0; d < 128; ++d)
      s += (qp[d] + b_in[hb + d]) * (kp[d] + b_in[512 + hb + d]);
    sc[nk] = s * 0.08838834764831845f;   // 1/sqrt(128)
  }
  float mx = fmaxf(fmaxf(sc[0], sc[1]), fmaxf(sc[2], sc[3]));
  float p[4], sum = 0.f;
#pragma unroll
  for (int nk = 0; nk < 4; ++nk) { p[nk] = __expf(sc[nk] - mx); sum += p[nk]; }
  float inv = 1.f / sum;
#pragma unroll
  for (int nk = 0; nk < 4; ++nk) p[nk] *= inv;
  float* op = ao + rows[nq] * D_MODEL + hb;
  for (int d = 0; d < 128; ++d) {
    float acc = 0.f;
#pragma unroll
    for (int nk = 0; nk < 4; ++nk)
      acc += p[nk] * (qkv[rows[nk] * 1536 + 1024 + hb + d] + b_in[1024 + hb + d]);
    op[d] = acc;
  }
}

// x = LayerNorm(x + o + bias) * g + bt   (in place over D_MODEL=512 rows)
__global__ void __launch_bounds__(256) resid_ln_kernel(
    float* __restrict__ x, const float* __restrict__ o, const float* __restrict__ bias,
    const float* __restrict__ g, const float* __restrict__ bt)
{
  __shared__ float sred[256];
  int r = blockIdx.x, t = threadIdx.x;
  float v0 = x[(size_t)r * 512 + t]       + o[(size_t)r * 512 + t]       + bias[t];
  float v1 = x[(size_t)r * 512 + t + 256] + o[(size_t)r * 512 + t + 256] + bias[t + 256];
  sred[t] = v0 + v1; __syncthreads();
  for (int st = 128; st > 0; st >>= 1) { if (t < st) sred[t] += sred[t + st]; __syncthreads(); }
  float mean = sred[0] * (1.f / 512.f); __syncthreads();
  float d0 = v0 - mean, d1 = v1 - mean;
  sred[t] = d0 * d0 + d1 * d1; __syncthreads();
  for (int st = 128; st > 0; st >>= 1) { if (t < st) sred[t] += sred[t + st]; __syncthreads(); }
  float inv = rsqrtf(sred[0] * (1.f / 512.f) + 1e-5f);
  x[(size_t)r * 512 + t]       = d0 * inv * g[t]       + bt[t];
  x[(size_t)r * 512 + t + 256] = d1 * inv * g[t + 256] + bt[t + 256];
}

// out_bf16[b*512+l, d] = bf16( LN( mean_n x[b,n,l,d] ) )
__global__ void __launch_bounds__(256) mean_ln_kernel(
    const float* __restrict__ x, const float* __restrict__ g,
    const float* __restrict__ bt, bf16* __restrict__ outb)
{
  __shared__ float sred[256];
  int r = blockIdx.x;                 // b*512 + l
  int b = r >> 9, l = r & 511;
  int t = threadIdx.x;
  float v[2];
#pragma unroll
  for (int i = 0; i < 2; ++i) {
    int d = t + i * 256;
    float s = 0.f;
#pragma unroll
    for (int n = 0; n < 4; ++n)
      s += x[((size_t)(b * 4 + n) * SEQ + l) * 512 + d];
    v[i] = 0.25f * s;
  }
  sred[t] = v[0] + v[1]; __syncthreads();
  for (int st = 128; st > 0; st >>= 1) { if (t < st) sred[t] += sred[t + st]; __syncthreads(); }
  float mean = sred[0] * (1.f / 512.f); __syncthreads();
  float d0 = v[0] - mean, d1 = v[1] - mean;
  sred[t] = d0 * d0 + d1 * d1; __syncthreads();
  for (int st = 128; st > 0; st >>= 1) { if (t < st) sred[t] += sred[t + st]; __syncthreads(); }
  float inv = rsqrtf(sred[0] * (1.f / 512.f) + 1e-5f);
  outb[(size_t)r * 512 + t]       = f2bf(d0 * inv * g[t]       + bt[t]);
  outb[(size_t)r * 512 + t + 256] = f2bf(d1 * inv * g[t + 256] + bt[t + 256]);
}

// ---------------------------------------------------------------------------
// Host-side orchestration
// ---------------------------------------------------------------------------
static void gemm(const bf16* A, const bf16* W, float* C, int M, int N, int K,
                 int accum, hipStream_t s) {
  dim3 grid(CEIL(N, 64), CEIL(M, 64));
  gemm_bf16_wmma<<<grid, 128, 0, s>>>(A, W, C, M, N, K, accum);
}
static void tobf(const float* src, bf16* dst, int n, hipStream_t s) {
  f2bf_kernel<<<CEIL(n, 256), 256, 0, s>>>(src, dst, n);
}

extern "C" void kernel_launch(void* const* d_in, const int* in_sizes, int n_in,
                              void* d_out, int out_size, void* d_ws, size_t ws_size,
                              hipStream_t stream) {
  (void)in_sizes; (void)n_in; (void)out_size; (void)ws_size;
  const int*   ids        = (const int*)  d_in[0];
  const float* emb        = (const float*)d_in[1];
  const float* in_proj_w  = (const float*)d_in[2];
  const float* conv_w     = (const float*)d_in[3];
  const float* conv_b     = (const float*)d_in[4];
  const float* x_proj_w   = (const float*)d_in[5];
  const float* dt_proj_w  = (const float*)d_in[6];
  const float* dt_proj_b  = (const float*)d_in[7];
  const float* a_log      = (const float*)d_in[8];
  const float* d_skip     = (const float*)d_in[9];
  const float* out_proj_w = (const float*)d_in[10];
  const float* attn_in_w  = (const float*)d_in[11];
  const float* attn_in_b  = (const float*)d_in[12];
  const float* attn_out_w = (const float*)d_in[13];
  const float* attn_out_b = (const float*)d_in[14];
  const float* attn_ln_g  = (const float*)d_in[15];
  const float* attn_ln_b  = (const float*)d_in[16];
  const float* normf_g    = (const float*)d_in[17];
  const float* normf_b    = (const float*)d_in[18];
  float* logits = (float*)d_out;

  // Workspace arena (aligned regions; aliased where lifetimes don't overlap).
  char* w = (char*)d_ws;
  size_t off = 0;
  auto alloc = [&](size_t bytes) -> void* {
    void* p = w + off;
    off = (off + bytes + 255) & ~(size_t)255;
    return p;
  };
  float* f_x     = (float*)alloc((size_t)NTOK * D_MODEL * 4);        //  8 MB residual stream
  float* f_xr    = (float*)alloc((size_t)NTOK * 2 * D_INNER * 4);    // 32 MB (alias: qkv, emb-bf16)
  float* f_xconv = (float*)alloc((size_t)NTOK * D_INNER * 4);        // 16 MB (alias: attn out ao)
  float* f_dt    = (float*)alloc((size_t)NTOK * D_INNER * 4);        // 16 MB (alias: attn proj tmp)
  float* f_y     = (float*)alloc((size_t)NTOK * D_INNER * 4);        // 16 MB
  float* f_dtbc  = (float*)alloc((size_t)NTOK * 64 * 4);             //  1 MB
  float* f_aneg  = (float*)alloc((size_t)D_INNER * D_STATE * 4);     // 64 KB
  bf16*  bb_act  = (bf16*)alloc((size_t)NTOK * D_INNER * 2);         //  8 MB activation staging
  bf16*  bb_w    = (bf16*)alloc((size_t)2 * D_INNER * D_MODEL * 2);  //  2 MB weight staging
  bf16*  bb_dt   = (bf16*)alloc((size_t)NTOK * DT_RANK * 2);         // 256 KB
  bf16*  bb_fin  = (bf16*)alloc((size_t)BATCH * SEQ * D_MODEL * 2);  //  1 MB
  float* f_qkv = f_xr;       // [4096, 1536] <= [4096, 2048]
  float* f_ao  = f_xconv;    // [4096, 512]
  float* f_tmp = f_dt;       // [4096, 512]
  bf16*  bb_emb = (bf16*)f_xr;  // 32.77 MB <= 33.55 MB

  // 0) Embed + broadcast to neurons.
  embed_kernel<<<CEIL(NTOK * D_MODEL, 256), 256, 0, stream>>>(ids, emb, f_x);

  for (int i = 0; i < 2; ++i) {
    const float* w_in   = in_proj_w  + (size_t)i * 2 * D_INNER * D_MODEL;
    const float* w_cv   = conv_w     + (size_t)i * 4 * D_INNER * 4;
    const float* b_cv   = conv_b     + (size_t)i * 4 * D_INNER;
    const float* w_xp   = x_proj_w   + (size_t)i * 64 * D_INNER;
    const float* w_dt   = dt_proj_w  + (size_t)i * D_INNER * DT_RANK;
    const float* b_dt   = dt_proj_b  + (size_t)i * D_INNER;
    const float* al     = a_log      + (size_t)i * D_INNER * D_STATE;
    const float* dsk    = d_skip     + (size_t)i * D_INNER;
    const float* w_out  = out_proj_w + (size_t)i * D_MODEL * D_INNER;
    const float* w_ai   = attn_in_w  + (size_t)i * 3 * D_MODEL * D_MODEL;
    const float* b_ai   = attn_in_b  + (size_t)i * 3 * D_MODEL;
    const float* w_ao   = attn_out_w + (size_t)i * D_MODEL * D_MODEL;
    const float* b_ao   = attn_out_b + (size_t)i * D_MODEL;
    const float* ln_g   = attn_ln_g  + (size_t)i * D_MODEL;
    const float* ln_b   = attn_ln_b  + (size_t)i * D_MODEL;

    // --- Mamba branch ---
    tobf(f_x, bb_act, NTOK * D_MODEL, stream);
    tobf(w_in, bb_w, 2 * D_INNER * D_MODEL, stream);
    gemm(bb_act, bb_w, f_xr, NTOK, 2 * D_INNER, D_MODEL, 0, stream);   // in_proj

    conv_silu_kernel<<<dim3(D_INNER / 256, BATCH * N_NEURONS), 256, 0, stream>>>(
        f_xr, w_cv, b_cv, f_xconv, bb_act);

    tobf(w_xp, bb_w, 64 * D_INNER, stream);
    gemm(bb_act, bb_w, f_dtbc, NTOK, 64, D_INNER, 0, stream);          // x_proj

    dtcols_kernel<<<CEIL(NTOK * DT_RANK, 256), 256, 0, stream>>>(f_dtbc, bb_dt, NTOK * DT_RANK);
    tobf(w_dt, bb_w, D_INNER * DT_RANK, stream);
    gemm(bb_dt, bb_w, f_dt, NTOK, D_INNER, DT_RANK, 0, stream);        // dt_proj
    softplus_kernel<<<CEIL(NTOK * D_INNER, 256), 256, 0, stream>>>(f_dt, b_dt, NTOK * D_INNER);

    negexp_kernel<<<CEIL(D_INNER * D_STATE, 256), 256, 0, stream>>>(al, f_aneg, D_INNER * D_STATE);
    ssm_scan_kernel<<<dim3(D_INNER / 256, BATCH * N_NEURONS), 256, 0, stream>>>(
        f_dt, f_xconv, f_dtbc, f_aneg, dsk, f_y);

    gate_kernel<<<CEIL(NTOK * D_INNER, 256), 256, 0, stream>>>(f_y, f_xr, bb_act, NTOK * D_INNER);
    tobf(w_out, bb_w, D_MODEL * D_INNER, stream);
    gemm(bb_act, bb_w, f_x, NTOK, D_MODEL, D_INNER, 1, stream);        // out_proj + residual

    // --- Attention over neurons ---
    tobf(f_x, bb_act, NTOK * D_MODEL, stream);
    tobf(w_ai, bb_w, 3 * D_MODEL * D_MODEL, stream);
    gemm(bb_act, bb_w, f_qkv, NTOK, 3 * D_MODEL, D_MODEL, 0, stream);  // qkv
    attn_kernel<<<CEIL(BATCH * SEQ * N_HEADS * N_NEURONS, 256), 256, 0, stream>>>(
        f_qkv, b_ai, f_ao);
    tobf(f_ao, bb_act, NTOK * D_MODEL, stream);
    tobf(w_ao, bb_w, D_MODEL * D_MODEL, stream);
    gemm(bb_act, bb_w, f_tmp, NTOK, D_MODEL, D_MODEL, 0, stream);      // attn out proj
    resid_ln_kernel<<<NTOK, 256, 0, stream>>>(f_x, f_tmp, b_ao, ln_g, ln_b);
  }

  // Final: neuron mean + LayerNorm -> bf16, tied-embedding logits GEMM.
  mean_ln_kernel<<<BATCH * SEQ, 256, 0, stream>>>(f_x, normf_g, normf_b, bb_fin);
  tobf(emb, bb_emb, VOCAB * D_MODEL, stream);
  gemm(bb_fin, bb_emb, logits, BATCH * SEQ, VOCAB, D_MODEL, 0, stream);
}
